// GraphEncoder_60129542800
// MI455X (gfx1250) — matrix operand order repaired and checked by reading, not compiled, not run on hardware
//
#include <hip/hip_runtime.h>
#include <hip/hip_bf16.h>

// ============================================================================
// DGCNN EdgeConv stack on MI455X (gfx1250, wave32, WMMA).
//
// Factoring: [x_i, x_j - x_i] @ W1 = At[i] + Ab[j] - Ab[i] with
//   At = X @ W1[:din],  Ab = X @ W1[din:]   (per-NODE GEMMs, 16x cheaper than per-edge)
// Per edge: relu(pre) @ W2 via 32-edge x 16-col WMMA tiles
// (v_wmma_f32_16x16x32_f16), f32-atomic accumulated at dst, then mean/BN/ReLU.
//
// Weights are pre-swizzled into WMMA-fragment-major layout (one contiguous
// 32B v16h per lane -> 2x global_load_b128), and both GEMM kernels use 2-way
// M register blocking so each B (weight) fragment feeds two WMMAs, halving
// the L2 weight-fetch bandwidth per MAC.
// ============================================================================

typedef __attribute__((ext_vector_type(16))) _Float16 v16h;
typedef __attribute__((ext_vector_type(8)))  _Float16 v8h;
typedef __attribute__((ext_vector_type(4)))  _Float16 v4h;
typedef __attribute__((ext_vector_type(8)))  float    v8f;
typedef __attribute__((ext_vector_type(4)))  float    v4f;

// ---------------------------------------------------------------------------
// WMMA fragment builders per CDNA5 ISA 7.12.2 (wave32).
// A (16x32 f16): lanes 0-15 & 16-31 hold rows M=0..15; lane-half selects K:
//   halves 0..7 -> K = half*8 + {0..7},  halves 8..15 -> K = 16+half*8+{0..7}
//   -> two contiguous 16B chunks per lane.
// B (32x16 f16): element e holds K = (e>>3)*16 + half*8 + (e&7), as A; N = lane&15 (pre-packed
//   so each lane's fragment is 32 contiguous bytes).
// C/D (16x16 f32): VGPR r -> row M = r + 8*half, N = lane&15.
// ---------------------------------------------------------------------------
__device__ inline v16h load_frag_a(const _Float16* __restrict__ Arow, int lane) {
  const int half = lane >> 4;
  const v8h lo = *(const v8h*)(Arow + half * 8);
  const v8h hi = *(const v8h*)(Arow + 16 + half * 8);
  v16h f;
#pragma unroll
  for (int i = 0; i < 8; ++i) { f[i] = lo[i]; f[8 + i] = hi[i]; }
  return f;
}

__device__ inline v16h load_frag_b_packed(const _Float16* __restrict__ Wf,
                                          int kchunks, int nt, int kc, int lane) {
  return *(const v16h*)(Wf + ((((size_t)nt * kchunks + kc) * 32 + lane) << 4));
}

// ---------------------------------------------------------------------------
// Utility kernels
// ---------------------------------------------------------------------------
__global__ void fill_f32(float* __restrict__ p, int n, float v) {
  int i = blockIdx.x * blockDim.x + threadIdx.x;
  if (i < n) p[i] = v;
}

// Pack rows [rowOff, rowOff+srcRows) of a row-major f32 [*, cols] matrix into
// WMMA-fragment-major f16: dst[((nt*kchunks + kc)*32 + lane)*16 + e], where
// element e of lane's fragment is B[k = kc*32 + (e>>3)*16 + half*8 + (e&7),
// n = nt*16 + (lane&15)]. K zero-padded to Kpad = kchunks*32.
__global__ void pack_w_frag(const float* __restrict__ W, _Float16* __restrict__ dst,
                            int srcRows, int cols, int rowOff, int kchunks) {
  const int i = blockIdx.x * blockDim.x + threadIdx.x;
  const int total = (cols >> 4) * kchunks * 512;
  if (i >= total) return;
  const int e    = i & 15;
  const int lane = (i >> 4) & 31;
  const int rest = i >> 9;
  const int kc   = rest % kchunks;
  const int nt   = rest / kchunks;
  const int half = lane >> 4, nl = lane & 15;
  const int k = kc * 32 + (e >> 3) * 16 + half * 8 + (e & 7);
  const int n = nt * 16 + nl;
  const float v = (k < srcRows) ? W[(size_t)(rowOff + k) * cols + n] : 0.0f;
  dst[i] = (_Float16)v;
}

// NeRF positional encoding of x[:, :9] (3 groups) + passthrough x[:, 9:16],
// packed to f16 with K padded (196 -> 224) for 32-wide WMMA K-steps.
__global__ void embed_pack_kernel(const float* __restrict__ x, _Float16* __restrict__ X16,
                                  int N, int ldx) {
  int n = blockIdx.x * blockDim.x + threadIdx.x;
  if (n >= N) return;
  const float* p = x + (size_t)n * 16;
  _Float16* o = X16 + (size_t)n * ldx;
  int w = 0;
#pragma unroll
  for (int g3 = 0; g3 < 3; ++g3) {
    const float p0 = p[3 * g3], p1 = p[3 * g3 + 1], p2 = p[3 * g3 + 2];
    o[w++] = (_Float16)p0; o[w++] = (_Float16)p1; o[w++] = (_Float16)p2;
    float f = 1.0f;
#pragma unroll
    for (int k = 0; k < 10; ++k) {
      o[w++] = (_Float16)__sinf(p0 * f);
      o[w++] = (_Float16)__sinf(p1 * f);
      o[w++] = (_Float16)__sinf(p2 * f);
      o[w++] = (_Float16)__cosf(p0 * f);
      o[w++] = (_Float16)__cosf(p1 * f);
      o[w++] = (_Float16)__cosf(p2 * f);
      f *= 2.0f;
    }
  }
#pragma unroll
  for (int c = 9; c < 16; ++c) o[w++] = (_Float16)p[c];
  for (; w < ldx; ++w) o[w] = (_Float16)0.0f;
}

__global__ void degree_kernel(const int* __restrict__ dstIdx, int E, float* __restrict__ cnt) {
  int e = blockIdx.x * blockDim.x + threadIdx.x;
  if (e < E) atomicAdd(&cnt[dstIdx[e]], 1.0f);
}

// ---------------------------------------------------------------------------
// Per-node GEMM: Y[N x Nout] f32 = X16[N x K] @ Wfrag[K x Nout]
// One wave per 32x16 output tile (2-way M blocking: each B fragment feeds
// two WMMAs); 8 waves per 256-thread block.
// ---------------------------------------------------------------------------
__global__ void gemm_node_kernel(const _Float16* __restrict__ X, int lda,
                                 const _Float16* __restrict__ Wf, int Nout,
                                 float* __restrict__ Y, int Nrows, int kchunks) {
  const int wave = threadIdx.x >> 5, lane = threadIdx.x & 31;
  const int half = lane >> 4, nl = lane & 15;
  const int ntiles = Nout >> 4;
  const int mtiles = (Nrows + 15) >> 4;
  const int mpairs = (mtiles + 1) >> 1;
  const int tile = blockIdx.x * 8 + wave;
  if (tile >= mpairs * ntiles) return;   // wave-uniform exit
  const int mp = tile / ntiles, nt = tile - mp * ntiles;

  int row0 = mp * 32 + nl;
  int row1 = row0 + 16;
  if (row0 >= Nrows) row0 = Nrows - 1;   // clamp (stores are guarded)
  if (row1 >= Nrows) row1 = Nrows - 1;
  const _Float16* A0 = X + (size_t)row0 * lda;
  const _Float16* A1 = X + (size_t)row1 * lda;

  v8f c0 = {}, c1 = {};
  for (int kc = 0; kc < kchunks; ++kc) {
    v16h b  = load_frag_b_packed(Wf, kchunks, nt, kc, lane);
    v16h a0 = load_frag_a(A0 + kc * 32, lane);
    v16h a1 = load_frag_a(A1 + kc * 32, lane);
    c0 = __builtin_amdgcn_wmma_f32_16x16x32_f16(false, a0, false, b, (short)0, c0,
                                                false, false);
    c1 = __builtin_amdgcn_wmma_f32_16x16x32_f16(false, a1, false, b, (short)0, c1,
                                                false, false);
  }
  const int n = nt * 16 + nl;
#pragma unroll
  for (int r = 0; r < 8; ++r) {
    const int m0 = mp * 32 + half * 8 + r;
    const int m1 = m0 + 16;
    if (m0 < Nrows) Y[(size_t)m0 * Nout + n] = c0[r];
    if (m1 < Nrows) Y[(size_t)m1 * Nout + n] = c1[r];
  }
}

// ---------------------------------------------------------------------------
// Edge kernel: one 256-thread block per 32-edge tile (2 WMMA M-subtiles).
// Phase 1: pre = relu(At[dst] + Ab[src] - Ab[dst] + b1) -> LDS f16 [32 x K2]
//          (float4 gathers, v4h LDS stores)
// Phase 2: each wave owns one n-tile; per K-chunk one B load feeds WMMAs for
//          both edge subtiles; scatter-accumulate rows into acc[dst] with
//          f32 atomics.
// ---------------------------------------------------------------------------
__global__ void edge_conv_kernel(const int* __restrict__ srcIdx, const int* __restrict__ dstIdx,
                                 int E,
                                 const float* __restrict__ At, const float* __restrict__ Ab,
                                 const float* __restrict__ b1, int K2, int kqShift,
                                 const _Float16* __restrict__ W2f, int dout,
                                 float* __restrict__ acc) {
  extern __shared__ _Float16 hbuf[];      // 32 * K2 halves
  __shared__ int ssrc[32], sdst[32];
  const int tid = threadIdx.x;
  const int e0 = blockIdx.x * 32;

  if (tid < 32) {
    int e = e0 + tid;
    if (e >= E) e = E - 1;
    ssrc[tid] = srcIdx[e];
    sdst[tid] = dstIdx[e];
  }
  __syncthreads();

  const int kq = K2 >> 2;                 // float4s per row (power of two)
  for (int idx = tid; idx < 32 * kq; idx += 256) {
    const int e = idx >> kqShift;
    const int k = (idx - (e << kqShift)) << 2;
    const v4f a  = *(const v4f*)(At + (size_t)sdst[e] * K2 + k);
    const v4f bs = *(const v4f*)(Ab + (size_t)ssrc[e] * K2 + k);
    const v4f bd = *(const v4f*)(Ab + (size_t)sdst[e] * K2 + k);
    const v4f bi = *(const v4f*)(b1 + k);
    v4h h;
#pragma unroll
    for (int j = 0; j < 4; ++j)
      h[j] = (_Float16)fmaxf(a[j] + bs[j] - bd[j] + bi[j], 0.0f);
    *(v4h*)(hbuf + (size_t)e * K2 + k) = h;
  }
  __syncthreads();

  const int wave = tid >> 5, lane = tid & 31;
  const int half = lane >> 4, nl = lane & 15;
  const int ntiles = dout >> 4;
  const int kchunks = K2 >> 5;
  const _Float16* A0 = hbuf + (size_t)nl * K2;          // edges 0..15
  const _Float16* A1 = hbuf + (size_t)(16 + nl) * K2;   // edges 16..31

  for (int nt = wave; nt < ntiles; nt += 8) {
    v8f c0 = {}, c1 = {};
    for (int kc = 0; kc < kchunks; ++kc) {
      v16h b  = load_frag_b_packed(W2f, kchunks, nt, kc, lane);
      v16h a0 = load_frag_a(A0 + kc * 32, lane);
      v16h a1 = load_frag_a(A1 + kc * 32, lane);
      c0 = __builtin_amdgcn_wmma_f32_16x16x32_f16(false, a0, false, b, (short)0, c0,
                                                  false, false);
      c1 = __builtin_amdgcn_wmma_f32_16x16x32_f16(false, a1, false, b, (short)0, c1,
                                                  false, false);
    }
    const int n = nt * 16 + nl;
#pragma unroll
    for (int r = 0; r < 8; ++r) {
      const int ea = half * 8 + r;            // D-row -> edge within tile
      const int eb = 16 + ea;
      if (e0 + ea < E) atomicAdd(&acc[(size_t)sdst[ea] * dout + n], c0[r]);
      if (e0 + eb < E) atomicAdd(&acc[(size_t)sdst[eb] * dout + n], c1[r]);
    }
  }
}

// ---------------------------------------------------------------------------
// Mean aggregation (+ second-MLP bias). Zero in-degree -> 0 (matches ref).
// ---------------------------------------------------------------------------
__global__ void finalize_kernel(const float* __restrict__ acc, const float* __restrict__ cnt,
                                const float* __restrict__ b2, float* __restrict__ out,
                                int N, int dout) {
  int i = blockIdx.x * blockDim.x + threadIdx.x;
  if (i >= N * dout) return;
  int n = i / dout, c = i - n * dout;
  float ct = cnt[n];
  out[i] = (ct > 0.0f) ? (acc[i] / ct + b2[c]) : 0.0f;
}

// Training-mode BatchNorm stats: one block per channel.
__global__ void bn_stats_kernel(const float* __restrict__ x, int N, int C,
                                float* __restrict__ mu, float* __restrict__ rs) {
  __shared__ float ssum[256], ssq[256];
  const int c = blockIdx.x;
  float s = 0.0f, q = 0.0f;
  for (int n = threadIdx.x; n < N; n += 256) {
    float v = x[(size_t)n * C + c];
    s += v; q += v * v;
  }
  ssum[threadIdx.x] = s; ssq[threadIdx.x] = q;
  __syncthreads();
  for (int st = 128; st > 0; st >>= 1) {
    if (threadIdx.x < st) {
      ssum[threadIdx.x] += ssum[threadIdx.x + st];
      ssq[threadIdx.x]  += ssq[threadIdx.x + st];
    }
    __syncthreads();
  }
  if (threadIdx.x == 0) {
    float m = ssum[0] / (float)N;
    float var = ssq[0] / (float)N - m * m;
    mu[c] = m;
    rs[c] = rsqrtf(var + 1e-5f);
  }
}

// BN affine + ReLU, repacked to f16 node features for the next layer's GEMMs.
__global__ void bn_apply_pack_kernel(const float* __restrict__ x,
                                     const float* __restrict__ mu, const float* __restrict__ rs,
                                     const float* __restrict__ g, const float* __restrict__ be,
                                     _Float16* __restrict__ X16, int N, int C) {
  int i = blockIdx.x * blockDim.x + threadIdx.x;
  if (i >= N * C) return;
  int n = i / C, c = i - n * C;
  float v = g[c] * (x[i] - mu[c]) * rs[c] + be[c];
  X16[(size_t)n * C + c] = (_Float16)fmaxf(v, 0.0f);
}

// ---------------------------------------------------------------------------
// Orchestration
// ---------------------------------------------------------------------------
extern "C" void kernel_launch(void* const* d_in, const int* in_sizes, int n_in,
                              void* d_out, int out_size, void* d_ws, size_t ws_size,
                              hipStream_t stream) {
  const float* x = (const float*)d_in[0];
  const int*  ei = (const int*)d_in[1];
  const int N = in_sizes[0] / 16;
  const int E = in_sizes[1] / 2;
  const int* srcIdx = ei;
  const int* dstIdx = ei + E;

  const float *Wa[5], *ba[5], *Wb[5], *bb[5];
  for (int l = 0; l < 5; ++l) {
    Wa[l] = (const float*)d_in[3 + 4 * l];
    ba[l] = (const float*)d_in[4 + 4 * l];
    Wb[l] = (const float*)d_in[5 + 4 * l];
    bb[l] = (const float*)d_in[6 + 4 * l];
  }
  const float *g[4], *be[4];
  for (int l = 0; l < 4; ++l) {
    g[l]  = (const float*)d_in[23 + 2 * l];
    be[l] = (const float*)d_in[24 + 2 * l];
  }

  const int din[5]  = {196, 64, 128, 256, 256};
  const int kpad[5] = {224, 64, 128, 256, 256};
  const int dout[5] = {64, 128, 256, 256, 512};

  char* ws = (char*)d_ws;
  size_t cur = 0;
  auto alloc = [&](size_t bytes) -> void* {
    void* p = ws + cur;
    cur += (bytes + 255) & ~(size_t)255;
    return p;
  };

  _Float16* X16 = (_Float16*)alloc((size_t)N * 256 * sizeof(_Float16));
  float* At  = (float*)alloc((size_t)N * 1024 * sizeof(float));
  float* Ab  = (float*)alloc((size_t)N * 1024 * sizeof(float));
  float* acc = (float*)alloc((size_t)N * 512 * sizeof(float));
  float* nf  = (float*)alloc((size_t)N * 256 * sizeof(float));
  float* cnt = (float*)alloc((size_t)N * sizeof(float));
  float* mu  = (float*)alloc(1024 * sizeof(float));
  float* rs  = (float*)alloc(1024 * sizeof(float));
  _Float16 *Wt16[5], *Wb16[5], *W2_16[5];
  for (int l = 0; l < 5; ++l) {
    const int K2 = 2 * dout[l];
    const size_t szA = (size_t)kpad[l] * K2 * sizeof(_Float16);   // kchunks*32 * Nout
    const size_t szB = (size_t)K2 * dout[l] * sizeof(_Float16);
    Wt16[l]  = (_Float16*)alloc(szA);
    Wb16[l]  = (_Float16*)alloc(szA);
    W2_16[l] = (_Float16*)alloc(szB);
  }

  auto cdiv = [](int a, int b) { return (a + b - 1) / b; };

  // Weight packing (f32 -> zero-padded, fragment-major f16)
  for (int l = 0; l < 5; ++l) {
    const int K2 = 2 * dout[l];
    const int kcA = kpad[l] / 32;
    const int totA = (K2 >> 4) * kcA * 512;
    pack_w_frag<<<cdiv(totA, 256), 256, 0, stream>>>(Wa[l], Wt16[l], din[l], K2, 0, kcA);
    pack_w_frag<<<cdiv(totA, 256), 256, 0, stream>>>(Wa[l], Wb16[l], din[l], K2, din[l], kcA);
    const int kcB = K2 / 32;
    const int totB = (dout[l] >> 4) * kcB * 512;
    pack_w_frag<<<cdiv(totB, 256), 256, 0, stream>>>(Wb[l], W2_16[l], K2, dout[l], 0, kcB);
  }

  // Positional encoding + in-degree (same graph all layers)
  embed_pack_kernel<<<cdiv(N, 128), 128, 0, stream>>>(x, X16, N, 224);
  fill_f32<<<cdiv(N, 256), 256, 0, stream>>>(cnt, N, 0.0f);
  degree_kernel<<<cdiv(E, 256), 256, 0, stream>>>(dstIdx, E, cnt);

  int lda = 224;
  for (int l = 0; l < 5; ++l) {
    const int K2 = 2 * dout[l];
    const int mtiles = cdiv(N, 16), ntiles = K2 / 16;
    const int mpairs = (mtiles + 1) / 2;
    const int tiles = mpairs * ntiles;
    const int kcA = kpad[l] / 32;

    // Per-node halves of the first edge-MLP GEMM
    gemm_node_kernel<<<cdiv(tiles, 8), 256, 0, stream>>>(X16, lda, Wt16[l], K2, At, N, kcA);
    gemm_node_kernel<<<cdiv(tiles, 8), 256, 0, stream>>>(X16, lda, Wb16[l], K2, Ab, N, kcA);

    fill_f32<<<cdiv(N * dout[l], 256), 256, 0, stream>>>(acc, N * dout[l], 0.0f);

    int kqShift = 0;
    while ((1 << (kqShift + 1)) <= (K2 >> 2)) ++kqShift;   // log2(K2/4)
    const size_t shmem = (size_t)32 * K2 * sizeof(_Float16);
    edge_conv_kernel<<<cdiv(E, 32), 256, shmem, stream>>>(srcIdx, dstIdx, E, At, Ab,
                                                          ba[l], K2, kqShift,
                                                          W2_16[l], dout[l], acc);

    if (l < 4) {
      finalize_kernel<<<cdiv(N * dout[l], 256), 256, 0, stream>>>(acc, cnt, bb[l], nf, N, dout[l]);
      bn_stats_kernel<<<dout[l], 256, 0, stream>>>(nf, N, dout[l], mu, rs);
      bn_apply_pack_kernel<<<cdiv(N * dout[l], 256), 256, 0, stream>>>(nf, mu, rs, g[l], be[l],
                                                                       X16, N, dout[l]);
      lda = dout[l];
    } else {
      finalize_kernel<<<cdiv(N * dout[l], 256), 256, 0, stream>>>(acc, cnt, bb[l],
                                                                  (float*)d_out, N, dout[l]);
    }
  }
}
